// RNN_Serial_28973849378893
// MI455X (gfx1250) — matrix operand order repaired and checked by reading, not compiled, not run on hardware
//
#include <hip/hip_runtime.h>

// Problem constants (match reference)
#define B_ 32
#define T_ 1024
#define H_ 1024
#define L_ 2
#define NBLK 64        // one block per 16 hidden columns: 64*16 = 1024
#define NTHREADS 256   // 8 wave32s

typedef __attribute__((ext_vector_type(16))) __bf16 v16bf;
typedef __attribute__((ext_vector_type(8)))  __bf16 v8bf;
typedef __attribute__((ext_vector_type(8)))  float  v8f;
typedef __attribute__((ext_vector_type(4)))  float  v4f;

// Force a 16-byte load on the GLOBAL path (addrspace(1)) no matter how the
// optimizer PHIs the pointer: avoids flat_load_b128 and its DScnt coupling.
__device__ __forceinline__ v8bf gload8(const __bf16* p) {
  typedef const v8bf __attribute__((address_space(1)))* gp_t;
  return *(gp_t)p;
}

// ---------------- prep kernels ----------------

__global__ void cvt_f32_bf16(const float* __restrict__ src, __bf16* __restrict__ dst, int n) {
  int i = blockIdx.x * blockDim.x + threadIdx.x;
  int stride = gridDim.x * blockDim.x;
  for (; i < n; i += stride) dst[i] = (__bf16)src[i];
}

__global__ void lstm_init(const float* __restrict__ h0, const float* __restrict__ c0,
                          const float* __restrict__ bih, const float* __restrict__ bhh,
                          float* __restrict__ biasC,
                          __bf16* __restrict__ h0buf,  // layer-0 hidden ping-pong [2,B,H]
                          __bf16* __restrict__ h1buf,  // layer-1 hidden ping-pong [2,B,H]
                          float* __restrict__ c_out, unsigned* __restrict__ bar) {
  int i = blockIdx.x * blockDim.x + threadIdx.x;
  if (i < L_ * B_ * H_) c_out[i] = c0[i];
  if (i < B_ * H_) {
    h0buf[i] = (__bf16)h0[i];                       // parity-0: read at phase 0
    h1buf[B_ * H_ + i] = (__bf16)h0[B_ * H_ + i];   // parity-1: read at phase 1
  }
  if (i < L_ * 4 * H_) biasC[i] = bih[i] + bhh[i];
  if (i == 0) *bar = 0u;
}

// ---------------- fragment loaders (CDNA5 WMMA VGPR layouts) ----------------

// A (16x32 bf16): lane<16 -> row=lane, K {kc..kc+7} then {kc+16..kc+23}
//                 lane>=16 -> row=lane-16, K {kc+8..kc+15} then {kc+24..kc+31}
__device__ __forceinline__ v16bf loadA_bf16(const __bf16* __restrict__ base, int rowStride,
                                            int row, int kc, int lane) {
  const __bf16* p = base + (size_t)row * rowStride + kc + ((lane >> 4) << 3);
  union { v16bf v; v8bf h[2]; } u;
  u.h[0] = gload8(p);
  u.h[1] = gload8(p + 16);
  return u.v;
}

// Same fragment but sourced from fp32 (x input), converted in-register to bf16.
__device__ __forceinline__ v16bf loadA_f32(const float* __restrict__ base, int rowStride,
                                           int row, int kc, int lane) {
  typedef const v4f __attribute__((address_space(1)))* gp_t;
  const float* p = base + (size_t)row * rowStride + kc + ((lane >> 4) << 3);
  v4f f0 = *(gp_t)(p);
  v4f f1 = *(gp_t)(p + 4);
  v4f f2 = *(gp_t)(p + 16);
  v4f f3 = *(gp_t)(p + 20);
  v16bf r;
#pragma unroll
  for (int i = 0; i < 4; ++i) {
    r[i]      = (__bf16)f0[i];
    r[4 + i]  = (__bf16)f1[i];
    r[8 + i]  = (__bf16)f2[i];
    r[12 + i] = (__bf16)f3[i];
  }
  return r;
}

// B (32x16 bf16, K x N): lane<16 -> col=lane, K {kc..kc+7} then {kc+16..kc+23}; lane>=16 -> col=lane-16, K {kc+8..kc+15} then {kc+24..kc+31}
// Source W is row-major [4H, H] (gates @ W^T), so a lane's 16 K-values lie in two contiguous runs of 8.
__device__ __forceinline__ v16bf loadB_bf16(const __bf16* __restrict__ W, int nBase,
                                            int kc, int lane) {
  const __bf16* p = W + (size_t)(nBase + (lane & 15)) * H_ + kc + ((lane >> 4) << 3);
  union { v16bf v; v8bf h[2]; } u;
  u.h[0] = gload8(p);
  u.h[1] = gload8(p + 16);
  return u.v;
}

// ---------------- grid barrier (monotonic counter, agent scope) ----------------

__device__ __forceinline__ void grid_sync(unsigned* __restrict__ bar, unsigned& gen) {
  __threadfence();
  __syncthreads();
  ++gen;
  if (threadIdx.x == 0) {
    unsigned target = gen * NBLK;
    __hip_atomic_fetch_add(bar, 1u, __ATOMIC_ACQ_REL, __HIP_MEMORY_SCOPE_AGENT);
    while (__hip_atomic_load(bar, __ATOMIC_ACQUIRE, __HIP_MEMORY_SCOPE_AGENT) < target) {
      __builtin_amdgcn_s_sleep(2);
    }
  }
  __syncthreads();
}

__device__ __forceinline__ float sigmoidf(float x) { return 1.0f / (1.0f + __expf(-x)); }

// ---------------- persistent LSTM kernel (layer-pipelined: 1 barrier/phase) ----

__global__ __launch_bounds__(NTHREADS) void lstm_persistent(
    const float*  __restrict__ x,        // [B,T,H] fp32
    const __bf16* __restrict__ Wih_bf,   // [L,4H,H]
    const __bf16* __restrict__ Whh_bf,   // [L,4H,H]
    const float*  __restrict__ biasC,    // [L,4H]  (bih+bhh)
    __bf16* __restrict__ h0buf,          // layer-0 hidden ping-pong [2,B,H]
    __bf16* __restrict__ h1buf,          // layer-1 hidden ping-pong [2,B,H]
    float*  __restrict__ h_out,          // [L,B,H] (d_out first half)
    float*  __restrict__ c_out,          // [L,B,H] (d_out second half, carries c)
    unsigned* __restrict__ bar) {

  __shared__ float lds[4 * B_ * 16];     // [gate][m][jl]  = 8 KB

  const int lane = threadIdx.x & 31;
  const int wv   = threadIdx.x >> 5;     // 8 waves
  const int gate = wv & 3;               // i,f,g,o
  const int mt   = wv >> 2;              // m-tile 0..1
  const int j0   = blockIdx.x * 16;      // hidden column base for this block
  const int mrow = mt * 16 + (lane & 15);
  const int nGlb = gate * H_ + j0;       // row base into W ([4H, H])
  const size_t hStride = (size_t)B_ * H_;

  unsigned gen = 0;

  // One layer-step: gates = bias + inp @ Wih^T + h_prev @ Whh^T, then cell update.
  auto process = [&](const float* xA, const __bf16* bfA,      // input (exactly one non-null)
                     const __bf16* Wih_l, const __bf16* Whh_l,
                     const __bf16* hprev,                     // prev hidden [B,H]
                     const float* biasL,                      // [4H]
                     __bf16* hw, float* cL, float* hL, bool writeH) {
    const float bv = biasL[nGlb + (lane & 15)];
    v8f acc = { bv, bv, bv, bv, bv, bv, bv, bv };

    if (xA) {
#pragma unroll 2
      for (int kc = 0; kc < H_; kc += 32) {
        v16bf a = loadA_f32(xA, T_ * H_, mrow, kc, lane);
        v16bf b = loadB_bf16(Wih_l, nGlb, kc, lane);
        acc = __builtin_amdgcn_wmma_f32_16x16x32_bf16(false, a, false, b,
                                                      (short)0, acc, false, false);
      }
    } else {
#pragma unroll 2
      for (int kc = 0; kc < H_; kc += 32) {
        v16bf a = loadA_bf16(bfA, H_, mrow, kc, lane);
        v16bf b = loadB_bf16(Wih_l, nGlb, kc, lane);
        acc = __builtin_amdgcn_wmma_f32_16x16x32_bf16(false, a, false, b,
                                                      (short)0, acc, false, false);
      }
    }
#pragma unroll 2
    for (int kc = 0; kc < H_; kc += 32) {
      v16bf a = loadA_bf16(hprev, H_, mrow, kc, lane);
      v16bf b = loadB_bf16(Whh_l, nGlb, kc, lane);
      acc = __builtin_amdgcn_wmma_f32_16x16x32_bf16(false, a, false, b,
                                                    (short)0, acc, false, false);
    }

    // C tile -> LDS: (r,lane) -> m = mt*16 + (lane>>4)*8 + r, jl = lane&15
    {
      const int mBase = mt * 16 + ((lane >> 4) << 3);
      const int jl = lane & 15;
#pragma unroll
      for (int r = 0; r < 8; ++r)
        lds[(gate * B_ + (mBase + r)) * 16 + jl] = acc[r];
    }
    __syncthreads();

    // Fused LSTM cell update: 512 elements, 2 per thread.
#pragma unroll
    for (int q = 0; q < 2; ++q) {
      const int e  = threadIdx.x + q * NTHREADS;   // 0..511
      const int m  = e >> 4;
      const int jl = e & 15;
      const float gi = lds[(0 * B_ + m) * 16 + jl];
      const float gf = lds[(1 * B_ + m) * 16 + jl];
      const float gg = lds[(2 * B_ + m) * 16 + jl];
      const float go = lds[(3 * B_ + m) * 16 + jl];
      const size_t idx = (size_t)m * H_ + j0 + jl;
      const float cprev = cL[idx];
      const float cn = sigmoidf(gf) * cprev + sigmoidf(gi) * tanhf(gg);
      const float hn = sigmoidf(go) * tanhf(cn);
      cL[idx] = cn;
      hw[idx] = (__bf16)hn;
      if (writeH) hL[idx] = hn;
    }
    __syncthreads();   // LDS reused by the next process() in this phase
  };

  const __bf16* Wih0 = Wih_bf;
  const __bf16* Wih1 = Wih_bf + (size_t)4 * H_ * H_;
  const __bf16* Whh0 = Whh_bf;
  const __bf16* Whh1 = Whh_bf + (size_t)4 * H_ * H_;

  // Phase p: layer 0 computes step p, layer 1 computes step p-1.
  // Both layers read parity-(p&1) hidden buffers (written in phase p-1),
  // and write parity-(1-(p&1)) buffers -> a single grid barrier per phase.
  for (int p = 0; p <= T_; ++p) {
    const size_t rdOff = (size_t)(p & 1) * hStride;
    const size_t wrOff = hStride - rdOff;
    __bf16* h0rd = h0buf + rdOff;
    __bf16* h0wr = h0buf + wrOff;
    __bf16* h1rd = h1buf + rdOff;
    __bf16* h1wr = h1buf + wrOff;

    if (p < T_) {
      // Layer 0, time step p: input = x[:,p,:], hidden = h0(p-1)
      process(x + (size_t)p * H_, nullptr, Wih0, Whh0, h0rd,
              biasC, h0wr, c_out, h_out, p == T_ - 1);
    }
    if (p >= 1) {
      // Layer 1, time step p-1: input = h0(p-1) (= h0rd), hidden = h1(p-2)
      process(nullptr, h0rd, Wih1, Whh1, h1rd,
              biasC + 4 * H_, h1wr, c_out + hStride,
              h_out + hStride, p == T_);
    }

    grid_sync(bar, gen);
  }
}

// ---------------- launch ----------------

extern "C" void kernel_launch(void* const* d_in, const int* in_sizes, int n_in,
                              void* d_out, int out_size, void* d_ws, size_t ws_size,
                              hipStream_t stream) {
  const float* x   = (const float*)d_in[0];   // [B,T,H]
  const float* h0  = (const float*)d_in[1];   // [L,B,H]
  const float* c0  = (const float*)d_in[2];   // [L,B,H]
  const float* Wih = (const float*)d_in[3];   // [L,4H,H]
  const float* Whh = (const float*)d_in[4];   // [L,4H,H]
  const float* bih = (const float*)d_in[5];   // [L,4H]
  const float* bhh = (const float*)d_in[6];   // [L,4H]

  float* h_out = (float*)d_out;                        // [L,B,H]
  float* c_out = (float*)d_out + (size_t)L_ * B_ * H_; // [L,B,H]

  // Workspace layout (~34 MB)
  char* ws = (char*)d_ws;
  const size_t wElems = (size_t)L_ * 4 * H_ * H_;      // 8,388,608 per matrix set
  const size_t hElems = (size_t)B_ * H_;               // 32,768 per hidden buffer
  __bf16* Wih_bf = (__bf16*)ws;  ws += wElems * sizeof(__bf16);
  __bf16* Whh_bf = (__bf16*)ws;  ws += wElems * sizeof(__bf16);
  __bf16* h0buf  = (__bf16*)ws;  ws += 2 * hElems * sizeof(__bf16);
  __bf16* h1buf  = (__bf16*)ws;  ws += 2 * hElems * sizeof(__bf16);
  float*  biasC  = (float*)ws;   ws += (size_t)L_ * 4 * H_ * sizeof(float);
  unsigned* bar  = (unsigned*)ws;

  // Prep: weights -> bf16 (once per call), bias sum, state init, barrier reset.
  cvt_f32_bf16<<<1024, 256, 0, stream>>>(Wih, Wih_bf, (int)wElems);
  cvt_f32_bf16<<<1024, 256, 0, stream>>>(Whh, Whh_bf, (int)wElems);
  lstm_init<<<(L_ * B_ * H_ + 255) / 256, 256, 0, stream>>>(h0, c0, bih, bhh,
                                                            biasC, h0buf, h1buf, c_out, bar);

  // Persistent layer-pipelined LSTM: 64 resident blocks, one grid barrier per phase.
  lstm_persistent<<<NBLK, NTHREADS, 0, stream>>>(x, Wih_bf, Whh_bf, biasC,
                                                 h0buf, h1buf,
                                                 h_out, c_out, bar);
}
